// BidirectionalMambaLayer_22840636080395
// MI455X (gfx1250) — matrix operand order repaired and checked by reading, not compiled, not run on hardware
//
#include <hip/hip_runtime.h>
#include <hip/hip_bf16.h>

// ---------------- problem constants ----------------
#define D_MODEL  768
#define D_STATE  16
#define D_INNER  1536           // 2*768
#define DT_RANK  48
#define DT_PAD   64             // pad K to multiple of 64 for the ping-pong K loop
#define D_CONV   4
#define BATCH    2
#define SEQ      1024
#define M_TOT    (BATCH*SEQ)    // 2048 rows in all GEMMs
#define XZ_W     (2*D_INNER)    // 3072
#define XDBL_W   (DT_RANK + 2*D_STATE)  // 80

#define GAS __attribute__((address_space(1)))   // global address space

typedef __bf16 bf16_t;
typedef __attribute__((ext_vector_type(16))) __bf16 v16bf;
typedef __attribute__((ext_vector_type(8)))  __bf16 v8bf;
typedef __attribute__((ext_vector_type(8)))  float  v8f;

__device__ __forceinline__ bf16_t f2bf(float f) {
    union { float f; unsigned u; } v; v.f = f;
    unsigned r = v.u + 0x7FFFu + ((v.u >> 16) & 1u);   // round-to-nearest-even
    unsigned short h = (unsigned short)(r >> 16);
    bf16_t out;
    __builtin_memcpy(&out, &h, 2);
    return out;
}
__device__ __forceinline__ float silu(float v) {
    return v * (1.0f / (1.0f + __expf(-v)));
}

// Globally-addressed fragment loads (force global_load_b128, LOADcnt only).
__device__ __forceinline__ v16bf load_b_frag(const bf16_t* p) {
    return *(const GAS v16bf*)p;
}
// A fragment loader: contiguous K order. Lanes 0-15 hold K = k0+[0..15];
// lanes 16-31 hold K = k0+[16..31] (matches B fragment).
// `p` already includes the lane's lh*16 K-offset; `off` selects the slab.
__device__ __forceinline__ v16bf load_a_frag(const bf16_t* p, int off) {
    v8bf lo = *(const GAS v8bf*)(p + off);
    v8bf hi = *(const GAS v8bf*)(p + off + 8);
    return __builtin_shufflevector(lo, hi, 0,1,2,3,4,5,6,7,8,9,10,11,12,13,14,15);
}

// ---------------- WMMA bf16 GEMM:  C[M,N] = A[M,K] * W[N,K]^T ----------------
// Wave computes a 32(M) x NT*16(N) tile; every B fragment feeds two WMMAs.
// K loop is manually unrolled 2x (ping-pong slabs of 32): each fragment
// register is defined once per iteration directly by a load (no rotation
// moves), all loads use immediate offsets from pointers bumped by 64/iter.
// Branch-free inner loop; EXEC all-ones around WMMA.
// Requires M % 256 == 0 and K % 64 == 0 (callers guarantee).
template<int NT>
__global__ __launch_bounds__(256)
void gemm_wmma_kernel(const bf16_t* __restrict__ A, const bf16_t* __restrict__ W,
                      float* __restrict__ C, int M, int N, int K, int n_base) {
    const int lane = threadIdx.x & 31;
    const int wave = threadIdx.x >> 5;
    const int m0 = (blockIdx.y * 8 + wave) * 32;
    const int n0 = n_base + blockIdx.x * (NT * 16);
    if (m0 >= M) return;                        // wave-uniform
    const int lm = lane & 15;                   // row (A) / column (B) in tile
    const int lh = lane >> 4;                   // K-half selector

    v8f acc[2][NT];
    #pragma unroll
    for (int tm = 0; tm < 2; ++tm)
        #pragma unroll
        for (int t = 0; t < NT; ++t) acc[tm][t] = (v8f){};

    const bf16_t* pa0 = A + (size_t)(m0 + lm) * K + lh * 16;
    const bf16_t* pa1 = pa0 + (size_t)16 * K;
    const bf16_t* pw[NT];
    #pragma unroll
    for (int t = 0; t < NT; ++t)
        pw[t] = W + (size_t)(n0 + t * 16 + lm) * K + lh * 16;

    // prologue: even slab (k = 0)
    v16bf a0e = load_a_frag(pa0, 0);
    v16bf a1e = load_a_frag(pa1, 0);
    v16bf be[NT];
    #pragma unroll
    for (int t = 0; t < NT; ++t) be[t] = load_b_frag(pw[t]);

    for (int k0 = 0; k0 < K - 64; k0 += 64) {
        // odd slab (k0+32) loads overlap the even-slab WMMAs below
        v16bf a0o = load_a_frag(pa0, 32);
        v16bf a1o = load_a_frag(pa1, 32);
        v16bf bo[NT];
        #pragma unroll
        for (int t = 0; t < NT; ++t) bo[t] = load_b_frag(pw[t] + 32);
        __builtin_prefetch(pa0 + 128, 0, 1);         // global_prefetch_b8

        #pragma unroll
        for (int t = 0; t < NT; ++t) {
            acc[0][t] = __builtin_amdgcn_wmma_f32_16x16x32_bf16(
                            false, a0e, false, be[t], (short)0, acc[0][t], false, false);
            acc[1][t] = __builtin_amdgcn_wmma_f32_16x16x32_bf16(
                            false, a1e, false, be[t], (short)0, acc[1][t], false, false);
        }

        // next even slab (k0+64) loads overlap the odd-slab WMMAs below
        pa0 += 64; pa1 += 64;
        a0e = load_a_frag(pa0, 0);
        a1e = load_a_frag(pa1, 0);
        #pragma unroll
        for (int t = 0; t < NT; ++t) { pw[t] += 64; be[t] = load_b_frag(pw[t]); }

        #pragma unroll
        for (int t = 0; t < NT; ++t) {
            acc[0][t] = __builtin_amdgcn_wmma_f32_16x16x32_bf16(
                            false, a0o, false, bo[t], (short)0, acc[0][t], false, false);
            acc[1][t] = __builtin_amdgcn_wmma_f32_16x16x32_bf16(
                            false, a1o, false, bo[t], (short)0, acc[1][t], false, false);
        }
    }

    // tail: even slab (K-64) already resident; load odd slab (K-32)
    {
        v16bf a0o = load_a_frag(pa0, 32);
        v16bf a1o = load_a_frag(pa1, 32);
        v16bf bo[NT];
        #pragma unroll
        for (int t = 0; t < NT; ++t) bo[t] = load_b_frag(pw[t] + 32);

        #pragma unroll
        for (int t = 0; t < NT; ++t) {
            acc[0][t] = __builtin_amdgcn_wmma_f32_16x16x32_bf16(
                            false, a0e, false, be[t], (short)0, acc[0][t], false, false);
            acc[1][t] = __builtin_amdgcn_wmma_f32_16x16x32_bf16(
                            false, a1e, false, be[t], (short)0, acc[1][t], false, false);
        }
        #pragma unroll
        for (int t = 0; t < NT; ++t) {
            acc[0][t] = __builtin_amdgcn_wmma_f32_16x16x32_bf16(
                            false, a0o, false, bo[t], (short)0, acc[0][t], false, false);
            acc[1][t] = __builtin_amdgcn_wmma_f32_16x16x32_bf16(
                            false, a1o, false, bo[t], (short)0, acc[1][t], false, false);
        }
    }

    // C/D layout: VGPR r -> M = m0 + tm*16 + 8*lh + r ; column = n0 + t*16 + lm
    #pragma unroll
    for (int tm = 0; tm < 2; ++tm) {
        #pragma unroll
        for (int t = 0; t < NT; ++t) {
            GAS float* cp = (GAS float*)(C + (size_t)(m0 + tm * 16 + lh * 8) * N
                                           + n0 + t * 16 + lm);
            #pragma unroll
            for (int r = 0; r < 8; ++r) cp[(size_t)r * N] = acc[tm][t][r];
        }
    }
}

// ---------------- elementwise prep kernels ----------------

// dst[r, 0..dst_stride) = bf16(src[r*src_stride + c]) for c < cols else 0
__global__ void pad_convert_kernel(const float* __restrict__ src, bf16_t* __restrict__ dst,
                                   int rows, int cols, int src_stride, int dst_stride) {
    int idx = blockIdx.x * blockDim.x + threadIdx.x;
    int total = rows * dst_stride;
    if (idx >= total) return;
    int r = idx / dst_stride, c = idx % dst_stride;
    float v = (c < cols) ? src[(size_t)r * src_stride + c] : 0.0f;
    dst[idx] = f2bf(v);
}

// x [B,SEQ,D_MODEL] f32 -> bf16, optionally time-flipped per batch
__global__ void convert_x_kernel(const float* __restrict__ x, bf16_t* __restrict__ dst, int flip) {
    int idx = blockIdx.x * blockDim.x + threadIdx.x;
    if (idx >= BATCH * SEQ * D_MODEL) return;
    int c = idx % D_MODEL;
    int l = (idx / D_MODEL) % SEQ;
    int b = idx / (D_MODEL * SEQ);
    int ls = flip ? (SEQ - 1 - l) : l;
    dst[idx] = f2bf(x[((size_t)b * SEQ + ls) * D_MODEL + c]);
}

// ---------------- causal depthwise conv + bias + SiLU ----------------
// xz [B,SEQ,3072] -> u (f32 and bf16) [B,SEQ,1536]
__global__ void conv_silu_kernel(const float* __restrict__ xz, const float* __restrict__ cw,
                                 const float* __restrict__ cb, float* __restrict__ u_f32,
                                 bf16_t* __restrict__ u_bf) {
    int idx = blockIdx.x * blockDim.x + threadIdx.x;
    if (idx >= BATCH * SEQ * D_INNER) return;
    int d = idx % D_INNER;
    int l = (idx / D_INNER) % SEQ;
    int b = idx / (D_INNER * SEQ);
    const float* base = xz + (size_t)b * SEQ * XZ_W;
    float s = cb[d];
    #pragma unroll
    for (int j = 0; j < D_CONV; ++j) {
        int li = l - (D_CONV - 1) + j;
        if (li >= 0) s += cw[d * D_CONV + j] * base[(size_t)li * XZ_W + d];
    }
    float act = silu(s);
    u_f32[idx] = act;
    u_bf[idx]  = f2bf(act);
}

// ---------------- selective scan (fp32 recurrence, fused gate) ----------------
// one thread per (batch, channel); 16-state h[] lives in VGPRs.
__global__ void scan_kernel(const float* __restrict__ delta_pre, const float* __restrict__ dt_b,
                            const float* __restrict__ u_f32, const float* __restrict__ x_dbl,
                            const float* __restrict__ xz, const float* __restrict__ Alog,
                            const float* __restrict__ Dskip, bf16_t* __restrict__ y_bf) {
    int ch = blockIdx.x * blockDim.x + threadIdx.x;
    if (ch >= BATCH * D_INNER) return;
    int b = ch / D_INNER, d = ch % D_INNER;
    float Arow[D_STATE];
    #pragma unroll
    for (int n = 0; n < D_STATE; ++n) Arow[n] = -__expf(Alog[d * D_STATE + n]);
    float h[D_STATE];
    #pragma unroll
    for (int n = 0; n < D_STATE; ++n) h[n] = 0.0f;
    const float bias = dt_b[d];
    const float Dd   = Dskip[d];
    for (int l = 0; l < SEQ; ++l) {
        size_t m = (size_t)b * SEQ + l;
        float xv  = delta_pre[m * D_INNER + d] + bias;
        float dlt = (xv > 20.0f) ? xv : log1pf(__expf(xv));   // softplus
        float uu  = u_f32[m * D_INNER + d];
        float du  = dlt * uu;
        const float* Bv = x_dbl + m * XDBL_W + DT_RANK;
        const float* Cv = Bv + D_STATE;
        float y = 0.0f;
        #pragma unroll
        for (int n = 0; n < D_STATE; ++n) {
            float dA = __expf(dlt * Arow[n]);
            h[n] = dA * h[n] + du * Bv[n];
            y += h[n] * Cv[n];
        }
        y += uu * Dd;
        float zv = xz[m * XZ_W + D_INNER + d];
        y *= silu(zv);
        y_bf[m * D_INNER + d] = f2bf(y);
    }
}

// ---------------- final combine: out += time-flipped bwd ----------------
__global__ void combine_kernel(const float* __restrict__ bwd, float* __restrict__ out) {
    int idx = blockIdx.x * blockDim.x + threadIdx.x;
    if (idx >= BATCH * SEQ * D_MODEL) return;
    int o = idx % D_MODEL;
    int l = (idx / D_MODEL) % SEQ;
    int b = idx / (D_MODEL * SEQ);
    out[idx] += bwd[((size_t)b * SEQ + (SEQ - 1 - l)) * D_MODEL + o];
}

// ---------------- host orchestration ----------------
extern "C" void kernel_launch(void* const* d_in, const int* in_sizes, int n_in,
                              void* d_out, int out_size, void* d_ws, size_t ws_size,
                              hipStream_t stream) {
    (void)in_sizes; (void)n_in; (void)out_size; (void)ws_size;
    const float* x = (const float*)d_in[0];

    // scratch carving (reused across directions)
    size_t off = 0;
    auto carve = [&](size_t bytes) -> void* {
        void* p = (char*)d_ws + off;
        off += (bytes + 255) & ~(size_t)255;
        return p;
    };
    bf16_t* xA_bf     = (bf16_t*)carve((size_t)M_TOT * D_MODEL * 2);
    bf16_t* inw_bf    = (bf16_t*)carve((size_t)XZ_W * D_MODEL * 2);
    bf16_t* xproj_bf  = (bf16_t*)carve((size_t)XDBL_W * D_INNER * 2);
    bf16_t* dtw_bf    = (bf16_t*)carve((size_t)D_INNER * DT_PAD * 2);
    bf16_t* outw_bf   = (bf16_t*)carve((size_t)D_MODEL * D_INNER * 2);
    float*  xz        = (float*)carve((size_t)M_TOT * XZ_W * 4);
    float*  u_f32     = (float*)carve((size_t)M_TOT * D_INNER * 4);
    bf16_t* u_bf      = (bf16_t*)carve((size_t)M_TOT * D_INNER * 2);
    float*  x_dbl     = (float*)carve((size_t)M_TOT * XDBL_W * 4);
    bf16_t* dt_bf     = (bf16_t*)carve((size_t)M_TOT * DT_PAD * 2);
    float*  delta_pre = (float*)carve((size_t)M_TOT * D_INNER * 4);
    bf16_t* y_bf      = (bf16_t*)carve((size_t)M_TOT * D_INNER * 2);
    float*  obuf      = (float*)carve((size_t)M_TOT * D_MODEL * 4);

    const int TB = 256;
    const unsigned GY = M_TOT / 256;   // 8 row-blocks (32 rows per wave, 8 waves)
    auto blocks = [](long long n, int tb) { return (unsigned)((n + tb - 1) / tb); };

    for (int dir = 0; dir < 2; ++dir) {
        const int base = 1 + dir * 9;
        const float* in_w    = (const float*)d_in[base + 0];
        const float* conv_w  = (const float*)d_in[base + 1];
        const float* conv_b  = (const float*)d_in[base + 2];
        const float* xproj_w = (const float*)d_in[base + 3];
        const float* dt_w    = (const float*)d_in[base + 4];
        const float* dt_b    = (const float*)d_in[base + 5];
        const float* Alog    = (const float*)d_in[base + 6];
        const float* Dskip   = (const float*)d_in[base + 7];
        const float* out_w   = (const float*)d_in[base + 8];

        // 1. inputs / weights -> bf16 (dt path zero-padded K 48 -> 64)
        convert_x_kernel<<<blocks((long long)M_TOT * D_MODEL, TB), TB, 0, stream>>>(x, xA_bf, dir);
        pad_convert_kernel<<<blocks((long long)XZ_W * D_MODEL, TB), TB, 0, stream>>>(
            in_w, inw_bf, XZ_W, D_MODEL, D_MODEL, D_MODEL);
        pad_convert_kernel<<<blocks((long long)XDBL_W * D_INNER, TB), TB, 0, stream>>>(
            xproj_w, xproj_bf, XDBL_W, D_INNER, D_INNER, D_INNER);
        pad_convert_kernel<<<blocks((long long)D_INNER * DT_PAD, TB), TB, 0, stream>>>(
            dt_w, dtw_bf, D_INNER, DT_RANK, DT_RANK, DT_PAD);
        pad_convert_kernel<<<blocks((long long)D_MODEL * D_INNER, TB), TB, 0, stream>>>(
            out_w, outw_bf, D_MODEL, D_INNER, D_INNER, D_INNER);

        // 2. in_proj: xz[2048,3072] = xA[2048,768] * in_w^T
        gemm_wmma_kernel<4><<<dim3(XZ_W / 64, GY), 256, 0, stream>>>(
            xA_bf, inw_bf, xz, M_TOT, XZ_W, D_MODEL, 0);

        // 3. causal depthwise conv + SiLU
        conv_silu_kernel<<<blocks((long long)M_TOT * D_INNER, TB), TB, 0, stream>>>(
            xz, conv_w, conv_b, u_f32, u_bf);

        // 4. x_proj: x_dbl[2048,80] = u * xproj_w^T  (cols 0-63, then 64-79)
        gemm_wmma_kernel<4><<<dim3(1, GY), 256, 0, stream>>>(
            u_bf, xproj_bf, x_dbl, M_TOT, XDBL_W, D_INNER, 0);
        gemm_wmma_kernel<1><<<dim3(1, GY), 256, 0, stream>>>(
            u_bf, xproj_bf, x_dbl, M_TOT, XDBL_W, D_INNER, 64);

        // 5. dt rank-expand: delta_pre[2048,1536] = dt_pad[2048,64] * dt_w_pad^T
        pad_convert_kernel<<<blocks((long long)M_TOT * DT_PAD, TB), TB, 0, stream>>>(
            x_dbl, dt_bf, M_TOT, DT_RANK, XDBL_W, DT_PAD);
        gemm_wmma_kernel<4><<<dim3(D_INNER / 64, GY), 256, 0, stream>>>(
            dt_bf, dtw_bf, delta_pre, M_TOT, D_INNER, DT_PAD, 0);

        // 6. selective scan + D-skip + SiLU(z) gate -> y (bf16)
        scan_kernel<<<blocks((long long)BATCH * D_INNER, TB), TB, 0, stream>>>(
            delta_pre, dt_b, u_f32, x_dbl, xz, Alog, Dskip, y_bf);

        // 7. out_proj (fwd overwrites d_out fully; bwd -> scratch, then flip-add)
        float* dst = (dir == 0) ? (float*)d_out : obuf;
        gemm_wmma_kernel<4><<<dim3(D_MODEL / 64, GY), 256, 0, stream>>>(
            y_bf, outw_bf, dst, M_TOT, D_MODEL, D_INNER, 0);
    }

    combine_kernel<<<blocks((long long)M_TOT * D_MODEL, TB), TB, 0, stream>>>(
        obuf, (float*)d_out);
}
